// LSTMCellModel_10746008175106
// MI455X (gfx1250) — matrix operand order repaired and checked by reading, not compiled, not run on hardware
//
#include <hip/hip_runtime.h>
#include <hip/hip_bf16.h>

typedef __attribute__((ext_vector_type(16))) __bf16 v16bf;
typedef __attribute__((ext_vector_type(8)))  float  v8f;

#define TSEQ 65536
#define IDIM 64
#define HDIM 512
#define NWG  32
#define NTHREADS 256
#define UNITS_PER_WG 16            // HDIM / NWG

// ---- LDS layout: per-wave WMMA A-fragment weight banks (bf16) ----
#define CHUNK_BYTES 1024           // 32 lanes * 32 B (one 16x32 bf16 A-tile)
#define L1_CHUNKS 18               // 2 x W_ih1 (K=64) + 16 x W_hh1 (K=512)
#define L2_CHUNKS 32               // 16 x W_ih2 + 16 x W_hh2 (K=1024 total)
#define L1_WAVE_BYTES (L1_CHUNKS * CHUNK_BYTES)
#define L2_WAVE_BYTES (L2_CHUNKS * CHUNK_BYTES)
#define OFF_L1  0
#define OFF_L2  (4 * L1_WAVE_BYTES)            // 73728
#define OFF_H1P (OFF_L2 + 4 * L2_WAVE_BYTES)   // 204800
#define OFF_H1C (OFF_H1P + 1024)
#define OFF_H2P (OFF_H1C + 1024)
#define OFF_X   (OFF_H2P + 1024)               // x_t as bf16[64] (128 B used)
#define OFF_G   (OFF_X + 256)                  // gate staging: 8 quadrants x 16 f32
#define OFF_C1  (OFF_G + 512)
#define OFF_C2  (OFF_C1 + 64)
#define SMEM_BYTES (OFF_C2 + 64)               // 208768 B < 320 KB LDS/WGP

// ---- workspace layout (bytes) ----
#define WS_H1  0        // bf16 [2][512] double-buffered
#define WS_H2  2048     // bf16 [2][512]
#define WS_H2F 4096     // f32  [512]
#define WS_CNT 6144     // barrier arrive counter
#define WS_GEN 6272     // barrier generation
#define WS_INIT_INTS 1600

#if __has_builtin(__builtin_amdgcn_tanhf)
__device__ inline float fast_tanh(float x) { return __builtin_amdgcn_tanhf(x); }
#else
__device__ inline float fast_tanh(float x) { return tanhf(x); }
#endif

union Frag { v16bf v; uint4 q[2]; };

__device__ inline float sigmoidf(float x) { return 1.0f / (1.0f + __expf(-x)); }

__device__ inline v8f wmma_bf16(v16bf a, v16bf b, v8f c) {
  return __builtin_amdgcn_wmma_f32_16x16x32_bf16(false, a, false, b, (short)0, c,
                                                 false, false);
}

// A-fragment: 32 B per lane, pre-packed in LDS
__device__ inline v16bf load_afrag(const char* base, int chunk, int lane) {
  const uint4* p = (const uint4*)(base + chunk * CHUNK_BYTES + lane * 32);
  Frag f; f.q[0] = p[0]; f.q[1] = p[1];
  return f.v;
}

// B-fragment: broadcast bf16 vector across all 16 columns.
// This code's B k order (contiguous, matches pack_w_chunk): lanes 0-15 cover K=koff..koff+15, lanes 16-31 K=+16.
__device__ inline v16bf bfrag_h(const __bf16* h, int koff, int lane) {
  const uint4* p = (const uint4*)(h + koff + ((lane >> 4) & 1) * 16);
  Frag f; f.q[0] = p[0]; f.q[1] = p[1];
  return f.v;
}

// Pack one 16x32 bf16 A-tile (rows row0..row0+15, cols koff..koff+31 of fp32 W)
// into this file's contiguous-K lane layout: lane m=L&15; lanes<16 hold K 0-15, lanes>=16
// hold K 16-31 (relative to koff).
__device__ inline void pack_w_chunk(char* base, int chunk, int lane,
                                    const float* __restrict__ W, int ld,
                                    int row0, int koff) {
  int m  = lane & 15;
  int kb = koff + ((lane >> 4) & 1) * 16;
  const float* src = W + (size_t)(row0 + m) * ld + kb;
  __bf16* dst = (__bf16*)(base + chunk * CHUNK_BYTES + lane * 32);
#pragma unroll
  for (int j = 0; j < 8; ++j) {
    dst[j]     = (__bf16)src[j];
    dst[8 + j] = (__bf16)src[8 + j];
  }
}

__device__ inline void grid_barrier(int* cnt, int* gen, int& mygen) {
  __syncthreads();
  if (threadIdx.x == 0) {
    int a = __hip_atomic_fetch_add(cnt, 1, __ATOMIC_ACQ_REL, __HIP_MEMORY_SCOPE_AGENT);
    if (a == NWG - 1) {
      __hip_atomic_store(cnt, 0, __ATOMIC_RELAXED, __HIP_MEMORY_SCOPE_AGENT);
      __hip_atomic_fetch_add(gen, 1, __ATOMIC_ACQ_REL, __HIP_MEMORY_SCOPE_AGENT);
    } else {
      while (__hip_atomic_load(gen, __ATOMIC_ACQUIRE, __HIP_MEMORY_SCOPE_AGENT) == mygen)
        __builtin_amdgcn_s_sleep(1);
    }
    ++mygen;
  }
  __syncthreads();
}

__global__ void init_ws_kernel(int* ws) {
  int i = blockIdx.x * blockDim.x + threadIdx.x;
  if (i < WS_INIT_INTS) ws[i] = 0;
}

__global__ __launch_bounds__(NTHREADS) void lstm_persistent_kernel(
    const float* __restrict__ x_seq,
    const float* __restrict__ Wih1, const float* __restrict__ Whh1,
    const float* __restrict__ bih1, const float* __restrict__ bhh1,
    const float* __restrict__ Wih2, const float* __restrict__ Whh2,
    const float* __restrict__ bih2, const float* __restrict__ bhh2,
    const float* __restrict__ Wout, const float* __restrict__ bout,
    float* __restrict__ out, char* __restrict__ ws) {
  extern __shared__ char smem[];
  const int tid  = threadIdx.x;
  const int lane = tid & 31;
  const int wave = tid >> 5;
  const int u0   = blockIdx.x * UNITS_PER_WG;

  __bf16* h1buf = (__bf16*)(ws + WS_H1);
  __bf16* h2buf = (__bf16*)(ws + WS_H2);
  float*  h2f   = (float*)(ws + WS_H2F);
  int* cnt = (int*)(ws + WS_CNT);
  int* gen = (int*)(ws + WS_GEN);

  __bf16* lds_h1p = (__bf16*)(smem + OFF_H1P);
  __bf16* lds_h1c = (__bf16*)(smem + OFF_H1C);
  __bf16* lds_h2p = (__bf16*)(smem + OFF_H2P);
  __bf16* lds_x   = (__bf16*)(smem + OFF_X);
  float*  lds_g   = (float*)(smem + OFF_G);    // [8][16]
  float*  lds_c1  = (float*)(smem + OFF_C1);
  float*  lds_c2  = (float*)(smem + OFF_C2);

  // ---- one-time: pack this WG's weight slice (bf16) into LDS; build bias C-frag ----
  char* wbase;
  v8f bias = {0, 0, 0, 0, 0, 0, 0, 0};
  if (wave < 4) {                        // layer-1 gate quadrant `wave`
    wbase = smem + OFF_L1 + wave * L1_WAVE_BYTES;
    int row0 = wave * HDIM + u0;
    pack_w_chunk(wbase, 0, lane, Wih1, IDIM, row0, 0);
    pack_w_chunk(wbase, 1, lane, Wih1, IDIM, row0, 32);
#pragma unroll
    for (int c = 0; c < 16; ++c)
      pack_w_chunk(wbase, 2 + c, lane, Whh1, HDIM, row0, c * 32);
#pragma unroll
    for (int r = 0; r < 8; ++r) {        // C/D layout: lanes<16 -> M=r, lanes>=16 -> M=r+8
      int row = row0 + r + ((lane >> 4) & 1) * 8;
      bias[r] = bih1[row] + bhh1[row];
    }
  } else {                               // layer-2 gate quadrant `wave-4`
    int q = wave - 4;
    wbase = smem + OFF_L2 + q * L2_WAVE_BYTES;
    int row0 = q * HDIM + u0;
#pragma unroll
    for (int c = 0; c < 16; ++c)
      pack_w_chunk(wbase, c, lane, Wih2, HDIM, row0, c * 32);
#pragma unroll
    for (int c = 0; c < 16; ++c)
      pack_w_chunk(wbase, 16 + c, lane, Whh2, HDIM, row0, c * 32);
#pragma unroll
    for (int r = 0; r < 8; ++r) {
      int row = row0 + r + ((lane >> 4) & 1) * 8;
      bias[r] = bih2[row] + bhh2[row];
    }
  }
  if (tid < UNITS_PER_WG) { lds_c1[tid] = 0.0f; lds_c2[tid] = 0.0f; }
  __syncthreads();

  int mygen = 0;

  for (int t = 0; t < TSEQ; ++t) {
    const int cur = t & 1, prev = cur ^ 1;

    // (a) stage h1(t-1) and x_t (fp32 -> bf16 once) into LDS
    ((unsigned int*)lds_h1p)[tid] = ((const unsigned int*)(h1buf + prev * HDIM))[tid];
    if (tid < IDIM) lds_x[tid] = (__bf16)x_seq[(size_t)t * IDIM + tid];
    if (tid == 0 && t + 1 < TSEQ)
      __builtin_prefetch(x_seq + (size_t)(t + 1) * IDIM, 0, 0);
    __syncthreads();

    // (b) layer-1 gates: two independent WMMA accumulation chains (even/odd K)
    if (wave < 4) {
      v8f acc0 = bias;
      v8f acc1 = {0, 0, 0, 0, 0, 0, 0, 0};
      acc0 = wmma_bf16(load_afrag(wbase, 0, lane), bfrag_h(lds_x, 0, lane), acc0);
      acc1 = wmma_bf16(load_afrag(wbase, 1, lane), bfrag_h(lds_x, 32, lane), acc1);
#pragma unroll
      for (int c = 0; c < 16; c += 2) {
        acc0 = wmma_bf16(load_afrag(wbase, 2 + c, lane),
                         bfrag_h(lds_h1p, c * 32, lane), acc0);
        acc1 = wmma_bf16(load_afrag(wbase, 3 + c, lane),
                         bfrag_h(lds_h1p, (c + 1) * 32, lane), acc1);
      }
      v8f acc = acc0 + acc1;
      if ((lane & 15) == 0) {
        int mb = (lane >> 4) * 8;
#pragma unroll
        for (int r = 0; r < 8; ++r) lds_g[wave * 16 + mb + r] = acc[r];
      }
    }
    __syncthreads();

    // (c) update c1/h1 for owned units; publish h1(t) as bf16
    if (tid < UNITS_PER_WG) {
      float gi = lds_g[0 * 16 + tid], gf = lds_g[1 * 16 + tid];
      float gg = lds_g[2 * 16 + tid], go = lds_g[3 * 16 + tid];
      float c1 = sigmoidf(gf) * lds_c1[tid] + sigmoidf(gi) * fast_tanh(gg);
      lds_c1[tid] = c1;
      float h1 = sigmoidf(go) * fast_tanh(c1);
      h1buf[cur * HDIM + u0 + tid] = (__bf16)h1;
      __threadfence();
    }

    // (e) one device-wide barrier per step (double buffering removes the 2nd)
    grid_barrier(cnt, gen, mygen);

    // (f) stage h1(t) and h2(t-1)
    ((unsigned int*)lds_h1c)[tid] = ((const unsigned int*)(h1buf + cur * HDIM))[tid];
    ((unsigned int*)lds_h2p)[tid] = ((const unsigned int*)(h2buf + prev * HDIM))[tid];
    __syncthreads();

    // (g) layer-2 gates over K=1024 = [h1(t); h2(t-1)], dual chains
    if (wave >= 4) {
      v8f acc0 = bias;
      v8f acc1 = {0, 0, 0, 0, 0, 0, 0, 0};
#pragma unroll
      for (int c = 0; c < 16; c += 2) {
        acc0 = wmma_bf16(load_afrag(wbase, c, lane),
                         bfrag_h(lds_h1c, c * 32, lane), acc0);
        acc1 = wmma_bf16(load_afrag(wbase, c + 1, lane),
                         bfrag_h(lds_h1c, (c + 1) * 32, lane), acc1);
      }
#pragma unroll
      for (int c = 0; c < 16; c += 2) {
        acc0 = wmma_bf16(load_afrag(wbase, 16 + c, lane),
                         bfrag_h(lds_h2p, c * 32, lane), acc0);
        acc1 = wmma_bf16(load_afrag(wbase, 17 + c, lane),
                         bfrag_h(lds_h2p, (c + 1) * 32, lane), acc1);
      }
      v8f acc = acc0 + acc1;
      if ((lane & 15) == 0) {
        int mb = (lane >> 4) * 8;
#pragma unroll
        for (int r = 0; r < 8; ++r) lds_g[wave * 16 + mb + r] = acc[r];
      }
    }
    __syncthreads();

    // (h) update c2/h2; publish h2(t) bf16 (+ fp32 copy for the head)
    if (tid < UNITS_PER_WG) {
      float gi = lds_g[4 * 16 + tid], gf = lds_g[5 * 16 + tid];
      float gg = lds_g[6 * 16 + tid], go = lds_g[7 * 16 + tid];
      float c2 = sigmoidf(gf) * lds_c2[tid] + sigmoidf(gi) * fast_tanh(gg);
      lds_c2[tid] = c2;
      float h2 = sigmoidf(go) * fast_tanh(c2);
      h2buf[cur * HDIM + u0 + tid] = (__bf16)h2;
      h2f[u0 + tid] = h2;
      __threadfence();
    }
  }

  // final head: out = h2 . W_out + b_out (fp32), computed by WG 0
  grid_barrier(cnt, gen, mygen);
  if (blockIdx.x == 0) {
    float s = 0.0f;
    for (int k = tid; k < HDIM; k += NTHREADS) s += h2f[k] * Wout[k];
    float* red = (float*)smem;        // weights no longer needed
    red[tid] = s;
    __syncthreads();
    for (int off = NTHREADS / 2; off > 0; off >>= 1) {
      if (tid < off) red[tid] += red[tid + off];
      __syncthreads();
    }
    if (tid == 0) out[0] = red[0] + bout[0];
  }
}

extern "C" void kernel_launch(void* const* d_in, const int* in_sizes, int n_in,
                              void* d_out, int out_size, void* d_ws, size_t ws_size,
                              hipStream_t stream) {
  (void)in_sizes; (void)n_in; (void)out_size; (void)ws_size;
  const float* x_seq = (const float*)d_in[0];
  const float* Wih1  = (const float*)d_in[1];
  const float* Whh1  = (const float*)d_in[2];
  const float* bih1  = (const float*)d_in[3];
  const float* bhh1  = (const float*)d_in[4];
  const float* Wih2  = (const float*)d_in[5];
  const float* Whh2  = (const float*)d_in[6];
  const float* bih2  = (const float*)d_in[7];
  const float* bhh2  = (const float*)d_in[8];
  const float* Wout  = (const float*)d_in[9];
  const float* bout  = (const float*)d_in[10];

  init_ws_kernel<<<dim3((WS_INIT_INTS + 255) / 256), dim3(256), 0, stream>>>((int*)d_ws);
  lstm_persistent_kernel<<<dim3(NWG), dim3(NTHREADS), SMEM_BYTES, stream>>>(
      x_seq, Wih1, Whh1, bih1, bhh1, Wih2, Whh2, bih2, bhh2, Wout, bout,
      (float*)d_out, (char*)d_ws);
}
